// WeatherPrediction_47785806135665
// MI455X (gfx1250) — matrix operand order repaired and checked by reading, not compiled, not run on hardware
//
#include <hip/hip_runtime.h>
#include <hip/hip_bf16.h>

// ---------------------------------------------------------------------------
// CDNA5 (gfx1250) implementation: MLP (GEMM+PReLU+GEMM) -> GCN -> PReLU -> GCN
// GEMMs use v_wmma_f32_16x16x32_bf16 (bf16 operands, fp32 accumulate), with a
// register-staged, software-pipelined K loop. Scatter-add uses f32 atomics.
// ---------------------------------------------------------------------------

typedef __attribute__((ext_vector_type(16))) __bf16 v16bf;
typedef __attribute__((ext_vector_type(8)))  __bf16 v8bf;
typedef __attribute__((ext_vector_type(4)))  __bf16 v4bf;
typedef __attribute__((ext_vector_type(8)))  float  v8f;

#define GEMM_TM   128
#define GEMM_TN   64
#define GEMM_TK   32
#define APAD      40   // bf16 elements per A row in LDS (conflict-avoiding pad)
#define BPAD      40   // bf16 elements per B column in LDS

// A: [M,K] fp32 row-major.  Wt: [Nf,K] bf16 (pre-transposed weights).
// flags: bit0 = add bias, bit1 = PReLU(alpha)
__global__ __launch_bounds__(256)
void wmma_gemm_bf16(const float* __restrict__ A, const __bf16* __restrict__ Wt,
                    const float* __restrict__ bias, const float* __restrict__ alpha,
                    float* __restrict__ C, int M, int K, int Nf, int flags)
{
    __shared__ __align__(16) __bf16 As[GEMM_TM * APAD];
    __shared__ __align__(16) __bf16 Bs[GEMM_TN * BPAD];

    const int t    = threadIdx.x & 255;   // provable range [0,256)
    const int wave = t >> 5;
    const int lane = t & 31;
    const int half = lane >> 4;           // which 16-lane group
    const int l16  = lane & 15;

    const int row0 = blockIdx.x * GEMM_TM;
    const int col0 = blockIdx.y * GEMM_TN;

    // per-thread staging assignments (constant across K steps)
    const int ar0  = t >> 3;              // A rows: thread covers rows ar0 + {0,32,64,96}
    const int ak4  = (t & 7) * 4;         // A k-offset (float4)
    const int bn   = t >> 2;              // B column
    const int bk8  = (t & 3) * 8;         // B k-offset (8 bf16)

    v8f acc[4];
#pragma unroll
    for (int ci = 0; ci < 4; ++ci)
#pragma unroll
        for (int j = 0; j < 8; ++j) acc[ci][j] = 0.0f;

    float4 va[4];
    v8bf   vb;

    // Edge rows are address-clamped (not zeroed): padding rows of D are never
    // stored, and rows of D depend only on the same row of A.
    auto load_tile = [&](int k0) {
#pragma unroll
        for (int i = 0; i < 4; ++i) {
            int gr  = row0 + ar0 + i * 32;
            int grs = (gr < M) ? gr : (M - 1);
            va[i] = *(const float4*)(A + (long long)grs * K + (k0 + ak4));
        }
        vb = *(const v8bf*)(Wt + (long long)(col0 + bn) * K + (k0 + bk8));
    };

    load_tile(0);

    for (int k0 = 0; k0 < K; k0 += GEMM_TK) {
        // ---- commit staged registers to LDS ----
#pragma unroll
        for (int i = 0; i < 4; ++i) {
            v4bf b;
            b[0] = (__bf16)va[i].x; b[1] = (__bf16)va[i].y;
            b[2] = (__bf16)va[i].z; b[3] = (__bf16)va[i].w;
            *(v4bf*)(&As[(ar0 + i * 32) * APAD + ak4]) = b;
        }
        *(v8bf*)(&Bs[bn * BPAD + bk8]) = vb;
        __syncthreads();

        // ---- prefetch next tile into registers (uniform branch, EXEC intact) ----
        if (k0 + GEMM_TK < K) load_tile(k0 + GEMM_TK);

        // ---- A fragment: 16x32 bf16 per wave ----
        // lane<16 : row=l16, K = 0..15
        // lane>=16: row=l16, K = 16..31
        v16bf afrag;
        {
            const __bf16* ar = &As[(wave * 16 + l16) * APAD];
#pragma unroll
            for (int j = 0; j < 8; ++j) {
                afrag[j]     = ar[half * 16 + j];
                afrag[j + 8] = ar[half * 16 + 8 + j];
            }
        }

        // ---- 4 column subtiles: B fragment 32x16; lane half selects K 0..15 / 16..31 ----
#pragma unroll
        for (int ci = 0; ci < 4; ++ci) {
            v16bf bfrag;
            const __bf16* br = &Bs[(ci * 16 + l16) * BPAD + half * 16];
#pragma unroll
            for (int j = 0; j < 16; ++j) bfrag[j] = br[j];

            acc[ci] = __builtin_amdgcn_wmma_f32_16x16x32_bf16(
                false, afrag, false, bfrag, (short)0, acc[ci], false, false);
        }
        __syncthreads();
    }

    // ---- epilogue: C/D layout: lane = N (mod 16), half selects M 0..7 / 8..15 ----
    const float al = (flags & 2) ? alpha[0] : 0.0f;
#pragma unroll
    for (int ci = 0; ci < 4; ++ci) {
        int gc = col0 + ci * 16 + l16;
        float bv = (flags & 1) ? bias[gc] : 0.0f;
#pragma unroll
        for (int r = 0; r < 8; ++r) {
            int gr = row0 + wave * 16 + half * 8 + r;
            if (gr < M) {
                float v = acc[ci][r] + bv;
                if (flags & 2) v = (v >= 0.0f) ? v : al * v;
                C[(long long)gr * Nf + gc] = v;
            }
        }
    }
}

// ---------------------------------------------------------------------------
// W [K,Nf] fp32 row-major  ->  Wt [Nf,K] bf16 (transposed)
__global__ void convert_transpose_kernel(const float* __restrict__ in, __bf16* __restrict__ out,
                                         int K, int Nf)
{
    int i = blockIdx.x * blockDim.x + threadIdx.x;
    if (i < K * Nf) {
        int k = i / Nf, n = i % Nf;
        out[n * K + k] = (__bf16)in[i];
    }
}

__global__ void fill_f32_kernel(float* __restrict__ p, float v, long long n)
{
    long long i = (long long)blockIdx.x * blockDim.x + threadIdx.x;
    if (i < n) p[i] = v;
}

__global__ void deg_count_kernel(const int* __restrict__ dst, float* __restrict__ deg, int E)
{
    int e = blockIdx.x * blockDim.x + threadIdx.x;
    if (e < E) atomicAdd(&deg[dst[e]], 1.0f);
}

__global__ void deg_rsqrt_kernel(const float* __restrict__ deg, float* __restrict__ dinv, int N)
{
    int i = blockIdx.x * blockDim.x + threadIdx.x;
    if (i < N) dinv[i] = rsqrtf(deg[i]);
}

// each thread: one edge x one float4 chunk of features, for one batch (blockIdx.y)
__global__ __launch_bounds__(256)
void scatter_edges_kernel(const float* __restrict__ hW, const int* __restrict__ src,
                          const int* __restrict__ dst, const float* __restrict__ dinv,
                          float* __restrict__ agg, int E, int Fv4, long long nodeStride)
{
    long long idx = (long long)blockIdx.x * blockDim.x + threadIdx.x;
    long long total = (long long)E * Fv4;
    if (idx >= total) return;
    int e = (int)(idx / Fv4);
    int c = (int)(idx % Fv4);
    int b = blockIdx.y;
    int s = src[e], d = dst[e];
    float nrm = dinv[s] * dinv[d];
    const float4* hp = (const float4*)(hW + (long long)b * nodeStride + (long long)s * (Fv4 * 4));
    float4 m = hp[c];
    float* ap = agg + (long long)b * nodeStride + (long long)d * (Fv4 * 4) + (long long)c * 4;
    atomicAdd(ap + 0, m.x * nrm);
    atomicAdd(ap + 1, m.y * nrm);
    atomicAdd(ap + 2, m.z * nrm);
    atomicAdd(ap + 3, m.w * nrm);
}

// out = agg + hW * dinv[node]^2 + bias   (+ optional PReLU)
__global__ __launch_bounds__(256)
void finalize_kernel(const float* __restrict__ agg, const float* __restrict__ hW,
                     const float* __restrict__ dinv, const float* __restrict__ bias,
                     const float* __restrict__ alpha, float* __restrict__ out,
                     int N, int F, long long total, int flags)
{
    long long idx = (long long)blockIdx.x * blockDim.x + threadIdx.x;
    if (idx >= total) return;
    int f = (int)(idx % F);
    long long nf = idx / F;
    int node = (int)(nf % N);
    float di = dinv[node];
    float v = agg[idx] + hW[idx] * (di * di) + bias[f];
    if (flags & 1) {
        float a = alpha[0];
        v = (v >= 0.0f) ? v : a * v;
    }
    out[idx] = v;
}

// ---------------------------------------------------------------------------
extern "C" void kernel_launch(void* const* d_in, const int* in_sizes, int n_in,
                              void* d_out, int out_size, void* d_ws, size_t ws_size,
                              hipStream_t stream)
{
    const float* X       = (const float*)d_in[0];
    const int*   edge    = (const int*)  d_in[1];
    const float* mlp_W1  = (const float*)d_in[2];
    const float* mlp_b1  = (const float*)d_in[3];
    const float* mlp_a1  = (const float*)d_in[4];
    const float* mlp_W2  = (const float*)d_in[5];
    const float* mlp_b2  = (const float*)d_in[6];
    const float* gcn_W1  = (const float*)d_in[7];
    const float* gcn_b1  = (const float*)d_in[8];
    const float* gcn_a   = (const float*)d_in[9];
    const float* gcn_W2  = (const float*)d_in[10];
    const float* gcn_b2  = (const float*)d_in[11];

    const int B = 2, F = 128, H = 256, OUT = 128;
    const int N = in_sizes[0] / (B * F);
    const int E = in_sizes[1] / 2;
    const int M = B * N;
    const int* src = edge;
    const int* dst = edge + E;

    // ---- workspace layout ----
    char* ws = (char*)d_ws;
    size_t off = 0;
    auto alloc = [&](size_t bytes) -> void* {
        off = (off + 255) & ~(size_t)255;
        void* p = ws + off;
        off += bytes;
        return p;
    };
    __bf16* wt_mlp1 = (__bf16*)alloc((size_t)F * H * sizeof(__bf16));    // [H, F=128]
    __bf16* wt_mlp2 = (__bf16*)alloc((size_t)H * H * sizeof(__bf16));    // [H, H]
    __bf16* wt_gcn1 = (__bf16*)alloc((size_t)H * H * sizeof(__bf16));    // [H, H]
    __bf16* wt_gcn2 = (__bf16*)alloc((size_t)H * OUT * sizeof(__bf16));  // [OUT, H]
    float*  deg     = (float*)alloc((size_t)N * sizeof(float));
    float*  dinv    = (float*)alloc((size_t)N * sizeof(float));
    float*  buf0    = (float*)alloc((size_t)M * H * sizeof(float));
    float*  buf1    = (float*)alloc((size_t)M * H * sizeof(float));
    float*  buf2    = (float*)alloc((size_t)M * H * sizeof(float));
    (void)ws_size; (void)n_in; (void)out_size;

    const dim3 blk(256);

    // ---- weights fp32 -> bf16, transposed to [Nf, K] ----
    convert_transpose_kernel<<<dim3((F * H + 255) / 256), blk, 0, stream>>>(mlp_W1, wt_mlp1, F, H);
    convert_transpose_kernel<<<dim3((H * H + 255) / 256), blk, 0, stream>>>(mlp_W2, wt_mlp2, H, H);
    convert_transpose_kernel<<<dim3((H * H + 255) / 256), blk, 0, stream>>>(gcn_W1, wt_gcn1, H, H);
    convert_transpose_kernel<<<dim3((H * OUT + 255) / 256), blk, 0, stream>>>(gcn_W2, wt_gcn2, H, OUT);

    // ---- degrees (self-loop included) ----
    fill_f32_kernel<<<dim3((N + 255) / 256), blk, 0, stream>>>(deg, 1.0f, (long long)N);
    deg_count_kernel<<<dim3((E + 255) / 256), blk, 0, stream>>>(dst, deg, E);
    deg_rsqrt_kernel<<<dim3((N + 255) / 256), blk, 0, stream>>>(deg, dinv, N);

    const int gmx = (M + GEMM_TM - 1) / GEMM_TM;

    // ---- MLP: Linear(128->256) + PReLU ----
    wmma_gemm_bf16<<<dim3(gmx, H / GEMM_TN), blk, 0, stream>>>(
        X, wt_mlp1, mlp_b1, mlp_a1, buf0, M, F, H, /*bias|prelu*/ 3);
    // ---- MLP: Linear(256->256) ----
    wmma_gemm_bf16<<<dim3(gmx, H / GEMM_TN), blk, 0, stream>>>(
        buf0, wt_mlp2, mlp_b2, nullptr, buf1, M, H, H, /*bias*/ 1);

    // ---- GCN layer 1: hW = h @ W1 (no bias yet) ----
    wmma_gemm_bf16<<<dim3(gmx, H / GEMM_TN), blk, 0, stream>>>(
        buf1, wt_gcn1, nullptr, nullptr, buf0, M, H, H, 0);
    // agg = 0
    {
        long long tot = (long long)M * H;
        fill_f32_kernel<<<dim3((unsigned)((tot + 255) / 256)), blk, 0, stream>>>(buf2, 0.0f, tot);
    }
    // scatter messages
    {
        int Fv4 = H / 4;
        long long tot = (long long)E * Fv4;
        scatter_edges_kernel<<<dim3((unsigned)((tot + 255) / 256), B), blk, 0, stream>>>(
            buf0, src, dst, dinv, buf2, E, Fv4, (long long)N * H);
    }
    // out1 = PReLU(agg + hW*dinv^2 + b1)
    {
        long long tot = (long long)B * N * H;
        finalize_kernel<<<dim3((unsigned)((tot + 255) / 256)), blk, 0, stream>>>(
            buf2, buf0, dinv, gcn_b1, gcn_a, buf1, N, H, tot, /*prelu*/ 1);
    }

    // ---- GCN layer 2: hW = x1 @ W2 (256->128) ----
    wmma_gemm_bf16<<<dim3(gmx, OUT / GEMM_TN), blk, 0, stream>>>(
        buf1, wt_gcn2, nullptr, nullptr, buf0, M, H, OUT, 0);
    {
        long long tot = (long long)M * OUT;
        fill_f32_kernel<<<dim3((unsigned)((tot + 255) / 256)), blk, 0, stream>>>(buf2, 0.0f, tot);
    }
    {
        int Fv4 = OUT / 4;
        long long tot = (long long)E * Fv4;
        scatter_edges_kernel<<<dim3((unsigned)((tot + 255) / 256), B), blk, 0, stream>>>(
            buf0, src, dst, dinv, buf2, E, Fv4, (long long)N * OUT);
    }
    {
        long long tot = (long long)B * N * OUT;
        finalize_kernel<<<dim3((unsigned)((tot + 255) / 256)), blk, 0, stream>>>(
            buf2, buf0, dinv, gcn_b2, nullptr, (float*)d_out, N, OUT, tot, 0);
    }
}
